// MambaBlock_5076651344516
// MI455X (gfx1250) — compile-verified, matrix operand order checked by reading, not run on hardware
//
#include <hip/hip_runtime.h>
#include <hip/hip_bf16.h>

#define D_MODEL 1024
#define D_INNER 2048
#define D_STATE 64
#define SEQ     2048

typedef __attribute__((ext_vector_type(16))) __bf16 v16bf;
typedef __attribute__((ext_vector_type(8)))  float  v8f;

union BFBits { __bf16 b; unsigned short u; };

__device__ __forceinline__ unsigned short f2bf(float f) {
    BFBits x; x.b = (__bf16)f; return x.u;
}

// ---------------------------------------------------------------------------
// f32 -> bf16 pre-conversion (packed uint2 stores, 4 elements/thread).
// One-time pass so GEMM inner loops touch only bf16 (half the L2 traffic,
// zero cvt VALU work competing with WMMA issue).
// ---------------------------------------------------------------------------
__global__ __launch_bounds__(256) void cvt_bf16_kernel(
    const float* __restrict__ src, __bf16* __restrict__ dst, int n4)
{
    int i = blockIdx.x * 256 + threadIdx.x;
    if (i >= n4) return;
    const float4 f = ((const float4*)src)[i];
    uint2 p;
    p.x = (unsigned)f2bf(f.x) | ((unsigned)f2bf(f.y) << 16);
    p.y = (unsigned)f2bf(f.z) | ((unsigned)f2bf(f.w) << 16);
    ((uint2*)dst)[i] = p;
}

// ---------------------------------------------------------------------------
// GEMM: Out[m,n] = sum_k A[m,k] * W[n,k]   (bf16 activation x bf16 weight^T)
// WMMA 16x16x32 bf16 -> f32. Tile 128x128x32, 256 threads = 8 waves (4x2),
// each wave owns 32x64 = 2x4 WMMA tiles.
// Staging: double-buffered async global->LDS DMA (ASYNCcnt), no VGPR transit.
// EPI: 0 = plain store, 1 = +bias then softplus (dt_proj).
// ---------------------------------------------------------------------------
template <int EPI>
__global__ __launch_bounds__(256) void gemm_bf16_wmma_async(
    const __bf16* __restrict__ A, int lda,
    const __bf16* __restrict__ W, int ldw,
    const float* __restrict__ bias,
    float* __restrict__ Out, int ldo,
    int M, int N, int K)
{
    constexpr int BM = 128, BN = 128, BK = 32;
    constexpr int LDH = 40;               // bf16 units; 80B pitch = 20 banks
    __shared__ __bf16 lA[2][BM * LDH];
    __shared__ __bf16 lB[2][BN * LDH];

    const int tid  = threadIdx.x;
    const int lane = tid & 31;
    const int wid  = tid >> 5;
    const int wm   = wid & 3;             // wave row (0..3) -> 32 rows
    const int wn   = wid >> 2;            // wave col (0..1) -> 64 cols
    const int half = lane >> 4;
    const int mrow = lane & 15;
    const int m0   = blockIdx.x * BM;
    const int n0   = blockIdx.y * BN;

    // Async-stage one 128x32 bf16 tile pair. 512 16B-chunks per matrix,
    // 2 chunks/thread/matrix -> 4 ASYNCcnt ops per wave per tile.
    auto stage = [&](int buf, int k0) {
        #pragma unroll
        for (int i = 0; i < 2; ++i) {
            int c = tid + i * 256;              // chunk id 0..511
            int r = c >> 2, col = (c & 3) << 3; // row, bf16 col (0,8,16,24)
            const __bf16* ga = &A[(size_t)(m0 + r) * lda + k0 + col];
            unsigned la = (unsigned)(size_t)&lA[buf][r * LDH + col];
            asm volatile("global_load_async_to_lds_b128 %0, %1, off"
                         :: "v"(la), "v"(ga) : "memory");
        }
        #pragma unroll
        for (int i = 0; i < 2; ++i) {
            int c = tid + i * 256;
            int r = c >> 2, col = (c & 3) << 3;
            const __bf16* gw = &W[(size_t)(n0 + r) * ldw + k0 + col];
            unsigned lb = (unsigned)(size_t)&lB[buf][r * LDH + col];
            asm volatile("global_load_async_to_lds_b128 %0, %1, off"
                         :: "v"(lb), "v"(gw) : "memory");
        }
    };

    v8f acc[2][4] = {};

    stage(0, 0);
    const int nk = K / BK;
    for (int t = 0; t < nk; ++t) {
        const int cur = t & 1;
        if (t + 1 < nk) {
            stage(cur ^ 1, (t + 1) * BK);
            // current tile's 4 ops done; next tile's 4 may stay in flight
            asm volatile("s_wait_asynccnt 0x4" ::: "memory");
        } else {
            asm volatile("s_wait_asynccnt 0x0" ::: "memory");
        }
        __syncthreads();

        // Fragment gather per 16-bit WMMA layout:
        //   k(v) = (v>>2)*16 + half*8 + (v&3)*2, two consecutive bf16 / VGPR
        union Frag { v16bf v; unsigned u[8]; };
        Frag fa[2], fb[4];
        #pragma unroll
        for (int tm = 0; tm < 2; ++tm) {
            int r = wm * 32 + tm * 16 + mrow;
            #pragma unroll
            for (int v = 0; v < 8; ++v) {
                int k = ((v >> 2) << 4) + (half << 3) + ((v & 3) << 1);
                fa[tm].u[v] = *(const unsigned*)&lA[cur][r * LDH + k];
            }
        }
        #pragma unroll
        for (int tn = 0; tn < 4; ++tn) {
            int r = wn * 64 + tn * 16 + mrow;
            #pragma unroll
            for (int v = 0; v < 8; ++v) {
                int k = ((v >> 2) << 4) + (half << 3) + ((v & 3) << 1);
                fb[tn].u[v] = *(const unsigned*)&lB[cur][r * LDH + k];
            }
        }
        #pragma unroll
        for (int tm = 0; tm < 2; ++tm)
            #pragma unroll
            for (int tn = 0; tn < 4; ++tn)
                acc[tm][tn] = __builtin_amdgcn_wmma_f32_16x16x32_bf16(
                    false, fa[tm].v, false, fb[tn].v,
                    (short)0, acc[tm][tn], false, false);
        __syncthreads();   // all reads of `cur` done before t+1 overwrites it
    }

    // D layout: lane -> N = lane&15, VGPR v -> M = v + 8*half
    #pragma unroll
    for (int tm = 0; tm < 2; ++tm) {
        #pragma unroll
        for (int tn = 0; tn < 4; ++tn) {
            int gn = n0 + wn * 64 + tn * 16 + mrow;
            #pragma unroll
            for (int v = 0; v < 8; ++v) {
                int gm = m0 + wm * 32 + tm * 16 + v + half * 8;
                float val = acc[tm][tn][v];
                if (EPI == 1) {
                    val += bias[gn];
                    val = (val > 20.f) ? val : log1pf(__expf(val));  // softplus
                }
                Out[(size_t)gm * ldo + gn] = val;
            }
        }
    }
}

// ---------------------------------------------------------------------------
// Causal depthwise conv (width 4) + bias + SiLU. Emits f32 (for scan) and
// bf16 (GEMM operand) copies.
// ---------------------------------------------------------------------------
__global__ __launch_bounds__(256) void conv_silu_kernel(
    const float* __restrict__ xr, const float* __restrict__ cw,
    const float* __restrict__ cb, float* __restrict__ xconv,
    __bf16* __restrict__ xconv_bf)
{
    int d = blockIdx.x * blockDim.x + threadIdx.x;   // channel
    int s = blockIdx.y;                              // seq pos
    float w0 = cw[d * 4 + 0], w1 = cw[d * 4 + 1];
    float w2 = cw[d * 4 + 2], w3 = cw[d * 4 + 3];
    float acc = cb[d];
    if (s >= 3) acc = fmaf(w0, xr[(size_t)(s - 3) * (2 * D_INNER) + d], acc);
    if (s >= 2) acc = fmaf(w1, xr[(size_t)(s - 2) * (2 * D_INNER) + d], acc);
    if (s >= 1) acc = fmaf(w2, xr[(size_t)(s - 1) * (2 * D_INNER) + d], acc);
    acc = fmaf(w3, xr[(size_t)s * (2 * D_INNER) + d], acc);
    float sig = 1.f / (1.f + __expf(-acc));
    float v = acc * sig;                             // SiLU
    xconv[(size_t)s * D_INNER + d] = v;
    xconv_bf[(size_t)s * D_INNER + d] = (__bf16)v;
}

// ---------------------------------------------------------------------------
// Selective scan: 16 lanes/channel, 4 states/lane; shfl_xor width-16 reduce.
// 128 blocks x 256 threads = 2048 channels.
// ---------------------------------------------------------------------------
__global__ __launch_bounds__(256) void scan_kernel(
    const float* __restrict__ xconv, const float* __restrict__ delta,
    const float* __restrict__ bc,    const float* __restrict__ A_log,
    float* __restrict__ y)
{
    const int tid = threadIdx.x;
    const int g   = tid & 15;
    const int d   = blockIdx.x * 16 + (tid >> 4);

    float Ac[4], st[4] = {0.f, 0.f, 0.f, 0.f};
    const float4 alog = *(const float4*)&A_log[(size_t)d * D_STATE + g * 4];
    Ac[0] = -__expf(alog.x); Ac[1] = -__expf(alog.y);
    Ac[2] = -__expf(alog.z); Ac[3] = -__expf(alog.w);

    for (int s = 0; s < SEQ; ++s) {
        float dt = delta[(size_t)s * D_INNER + d];
        float xv = xconv[(size_t)s * D_INNER + d];
        const float4 Bv = *(const float4*)&bc[(size_t)s * (2 * D_STATE) + g * 4];
        const float4 Cv = *(const float4*)&bc[(size_t)s * (2 * D_STATE) + D_STATE + g * 4];
        float dx = dt * xv;
        float accv = 0.f;
        { float dA = __expf(dt * Ac[0]); st[0] = fmaf(dA, st[0], dx * Bv.x); accv = fmaf(st[0], Cv.x, accv); }
        { float dA = __expf(dt * Ac[1]); st[1] = fmaf(dA, st[1], dx * Bv.y); accv = fmaf(st[1], Cv.y, accv); }
        { float dA = __expf(dt * Ac[2]); st[2] = fmaf(dA, st[2], dx * Bv.z); accv = fmaf(st[2], Cv.z, accv); }
        { float dA = __expf(dt * Ac[3]); st[3] = fmaf(dA, st[3], dx * Bv.w); accv = fmaf(st[3], Cv.w, accv); }
        accv += __shfl_xor(accv, 1, 16);
        accv += __shfl_xor(accv, 2, 16);
        accv += __shfl_xor(accv, 4, 16);
        accv += __shfl_xor(accv, 8, 16);
        if (g == 0) y[(size_t)s * D_INNER + d] = accv;
    }
}

// ---------------------------------------------------------------------------
// Gate: ymod_bf = bf16( y * silu(res) ), res = xr[:, D_INNER:2*D_INNER]
// ---------------------------------------------------------------------------
__global__ __launch_bounds__(256) void gate_kernel(
    const float* __restrict__ y, const float* __restrict__ xr,
    __bf16* __restrict__ ymod_bf)
{
    int i = blockIdx.x * 256 + threadIdx.x;
    int s = i >> 11, dcol = i & (D_INNER - 1);
    float r = xr[(size_t)s * (2 * D_INNER) + D_INNER + dcol];
    float sig = 1.f / (1.f + __expf(-r));
    ymod_bf[i] = (__bf16)(y[i] * (r * sig));
}

// ---------------------------------------------------------------------------
extern "C" void kernel_launch(void* const* d_in, const int* in_sizes, int n_in,
                              void* d_out, int out_size, void* d_ws, size_t ws_size,
                              hipStream_t stream)
{
    (void)in_sizes; (void)n_in; (void)out_size; (void)ws_size;
    const float* hs    = (const float*)d_in[0];   // (2048, 1024)
    const float* w_in  = (const float*)d_in[1];   // (4096, 1024)
    const float* cw    = (const float*)d_in[2];   // (2048, 1, 4)
    const float* cb    = (const float*)d_in[3];   // (2048,)
    const float* w_x   = (const float*)d_in[4];   // (128, 2048)
    const float* w_dt  = (const float*)d_in[5];   // (2048, 2048)
    const float* b_dt  = (const float*)d_in[6];   // (2048,)
    const float* A_log = (const float*)d_in[7];   // (2048, 64)
    const float* w_out = (const float*)d_in[8];   // (1024, 2048)
    float* out = (float*)d_out;                   // (2048, 1024) fp32

    char* ws = (char*)d_ws;
    float*  xr      = (float*)(ws);                    // 2048x4096  f32
    float*  xcv     = (float*)(ws + 33554432);         // 2048x2048  f32
    float*  dlt     = (float*)(ws + 50331648);         // 2048x2048  f32
    float*  yv      = (float*)(ws + 67108864);         // 2048x2048  f32
    float*  bcm     = (float*)(ws + 83886080);         // 2048x128   f32
    __bf16* hs_bf   = (__bf16*)(ws + 84934656);        // 2048x1024
    __bf16* w_in_bf = (__bf16*)(ws + 89128960);        // 4096x1024
    __bf16* w_x_bf  = (__bf16*)(ws + 97517568);        // 128x2048
    __bf16* w_dt_bf = (__bf16*)(ws + 98041856);        // 2048x2048
    __bf16* w_out_bf= (__bf16*)(ws + 106430464);       // 1024x2048
    __bf16* xcv_bf  = (__bf16*)(ws + 110624768);       // 2048x2048
    __bf16* ymod_bf = xcv_bf;                          // reuse after scan

    dim3 blk(256);

    // 0) one-time f32 -> bf16 conversions (weights + hidden states)
    cvt_bf16_kernel<<<dim3((SEQ * D_MODEL / 4) / 256), blk, 0, stream>>>(hs, hs_bf, SEQ * D_MODEL / 4);
    cvt_bf16_kernel<<<dim3((2 * D_INNER * D_MODEL / 4) / 256), blk, 0, stream>>>(w_in, w_in_bf, 2 * D_INNER * D_MODEL / 4);
    cvt_bf16_kernel<<<dim3((2 * D_STATE * D_INNER / 4) / 256), blk, 0, stream>>>(w_x, w_x_bf, 2 * D_STATE * D_INNER / 4);
    cvt_bf16_kernel<<<dim3((D_INNER * D_INNER / 4) / 256), blk, 0, stream>>>(w_dt, w_dt_bf, D_INNER * D_INNER / 4);
    cvt_bf16_kernel<<<dim3((D_MODEL * D_INNER / 4) / 256), blk, 0, stream>>>(w_out, w_out_bf, D_MODEL * D_INNER / 4);

    // 1) in_proj: xr = hs x in_proj_w^T           (2048 x 4096)
    gemm_bf16_wmma_async<0><<<dim3(SEQ / 128, (2 * D_INNER) / 128), blk, 0, stream>>>(
        hs_bf, D_MODEL, w_in_bf, D_MODEL, nullptr, xr, 2 * D_INNER, SEQ, 2 * D_INNER, D_MODEL);

    // 2) causal depthwise conv + bias + SiLU      (2048 x 2048, f32 + bf16)
    conv_silu_kernel<<<dim3(D_INNER / 256, SEQ), blk, 0, stream>>>(xr, cw, cb, xcv, xcv_bf);

    // 3) x_proj: bc = xcv x x_proj_w^T            (2048 x 128)
    gemm_bf16_wmma_async<0><<<dim3(SEQ / 128, 1), blk, 0, stream>>>(
        xcv_bf, D_INNER, w_x_bf, D_INNER, nullptr, bcm, 2 * D_STATE, SEQ, 2 * D_STATE, D_INNER);

    // 4) dt_proj + bias + softplus: delta         (2048 x 2048)
    gemm_bf16_wmma_async<1><<<dim3(SEQ / 128, D_INNER / 128), blk, 0, stream>>>(
        xcv_bf, D_INNER, w_dt_bf, D_INNER, b_dt, dlt, D_INNER, SEQ, D_INNER, D_INNER);

    // 5) selective scan -> y                      (2048 x 2048)
    scan_kernel<<<dim3(D_INNER / 16), blk, 0, stream>>>(xcv, dlt, bcm, A_log, yv);

    // 6) gate: ymod_bf = bf16( y * silu(res) )    (reuses xcv_bf storage)
    gate_kernel<<<dim3((SEQ * D_INNER) / 256), blk, 0, stream>>>(yv, xr, ymod_bf);

    // 7) out_proj: out = ymod x out_proj_w^T      (2048 x 1024)
    gemm_bf16_wmma_async<0><<<dim3(SEQ / 128, D_MODEL / 128), blk, 0, stream>>>(
        ymod_bf, D_INNER, w_out_bf, D_INNER, nullptr, out, D_MODEL, SEQ, D_MODEL, D_INNER);
}
